// CrossAttention_66571993088543
// MI455X (gfx1250) — matrix operand order repaired and checked by reading, not compiled, not run on hardware
//
#include <hip/hip_runtime.h>

// ---------------------------------------------------------------------------
// CrossAttention forward for MI455X (gfx1250, wave32, WMMA).
//   q = q_in @ Wq + bq ; k = k_v @ Wk + bk ; v = k_v @ Wv + bv   (bf16 in ws)
//   S = (q @ k^T) * scale (fp32 ws) ; P = softmax(S) ; out = P @ v (fp32)
// All GEMMs: v_wmma_f32_16x16x32_bf16, 128x128 block tile, 8 waves,
// K-chunk 64, double-buffered LDS software pipeline.
// qk_kernel stages its (already-bf16) tiles with GLOBAL_LOAD_ASYNC_TO_LDS_B128.
// Workspace: q(32MB bf16) | k(32MB) | v(32MB) | S(128MB fp32) = 224MB.
// ---------------------------------------------------------------------------

#define EMBED 1024
#define BATCH 8
#define SEQ_I 2048
#define SEQ_J 2048
#define QK_SCALE 0.03125f /* 1/sqrt(1024) */

typedef __attribute__((ext_vector_type(16))) __bf16 v16bf;
typedef __attribute__((ext_vector_type(8)))  float  v8f;

union Frag16 { v16bf v; uint4 q[2]; };
union Acc8   { v8f v; float f[8]; };

// fp32 -> bf16 round-to-nearest-even
__device__ __forceinline__ unsigned short f2bf(float x) {
    unsigned int u = __float_as_uint(x);
    u += 0x7fffu + ((u >> 16) & 1u);
    return (unsigned short)(u >> 16);
}

// LDS row stride in halfwords: 64 data + 8 pad = 144B/row.
// 144 % 16 == 0 (aligned b128), 36-bank shift/row, gcd(36,64)=4 -> 16 rows
// hit 16 distinct bank groups: conflict-free fragment reads.
#define LDS_STRIDE 72

// LDS byte offset of a shared-memory pointer: flat LDS aperture address
// truncates to the wave-relative LDS address (ISA 10.2).
__device__ __forceinline__ unsigned lds_off(const void* p) {
    return (unsigned)(size_t)p;
}

// Async 16B copy global -> LDS (per lane), tracked by ASYNCcnt.
__device__ __forceinline__ void async_copy_b128(unsigned lds_byte, const void* gptr) {
    asm volatile("global_load_async_to_lds_b128 %0, %1, off"
                 :: "v"(lds_byte), "v"((unsigned long long)(size_t)gptr)
                 : "memory");
}
__device__ __forceinline__ void wait_asynccnt0() {
    asm volatile("s_wait_asynccnt 0x0" ::: "memory");
}

// Wave micro-kernel: 64(M) x 32(N) x 64(K) using 16 WMMAs from one LDS buffer.
__device__ __forceinline__ void wmma_tile(const unsigned short* ldsA, const unsigned short* ldsB,
                                          int wm, int wn, int col, int kh, v8f acc[4][2]) {
#pragma unroll
    for (int ks = 0; ks < 64; ks += 32) {
        Frag16 af[4], bf[2];
#pragma unroll
        for (int mi = 0; mi < 4; ++mi) {
            const int r = (wm * 64 + mi * 16 + col) * LDS_STRIDE + ks;
            af[mi].q[0] = *(const uint4*)&ldsA[r + kh * 16];
            af[mi].q[1] = *(const uint4*)&ldsA[r + kh * 16 + 8];
        }
#pragma unroll
        for (int ni = 0; ni < 2; ++ni) {
            const int c = (wn * 32 + ni * 16 + col) * LDS_STRIDE + ks + kh * 16;
            bf[ni].q[0] = *(const uint4*)&ldsB[c];
            bf[ni].q[1] = *(const uint4*)&ldsB[c + 8];
        }
#pragma unroll
        for (int mi = 0; mi < 4; ++mi)
#pragma unroll
            for (int ni = 0; ni < 2; ++ni)
                acc[mi][ni] = __builtin_amdgcn_wmma_f32_16x16x32_bf16(
                    false, af[mi].v, false, bf[ni].v, (short)0, acc[mi][ni], false, false);
    }
}

// ---------------------------------------------------------------------------
// Y[M][1024](bf16) = X[M][1024](fp32) @ W[1024][1024](fp32) + bias
__global__ __launch_bounds__(256)
void proj_kernel(const float* __restrict__ X, const float* __restrict__ W,
                 const float* __restrict__ bias, unsigned short* __restrict__ Y) {
    __shared__ unsigned short lds_A[2][128 * LDS_STRIDE];
    __shared__ unsigned short lds_B[2][128 * LDS_STRIDE];

    const int tid  = threadIdx.x;
    const int n0   = blockIdx.x * 128;
    const int m0   = blockIdx.y * 128;
    const int w    = tid >> 5;
    const int lane = tid & 31;
    const int wm   = w >> 2;
    const int wn   = w & 3;
    const int col  = lane & 15;
    const int kh   = lane >> 4;

    auto stageA = [&](unsigned short* buf, int k0) {  // X[m0..+127][k0..+63] fp32->bf16
#pragma unroll
        for (int s = tid; s < 2048; s += 256) {
            const int row = s >> 4;
            const int c4  = (s & 15) << 2;
            const float4 f = *(const float4*)(X + (size_t)(m0 + row) * EMBED + k0 + c4);
            unsigned short* p = &buf[row * LDS_STRIDE + c4];
            p[0] = f2bf(f.x); p[1] = f2bf(f.y); p[2] = f2bf(f.z); p[3] = f2bf(f.w);
        }
    };
    auto stageB = [&](unsigned short* buf, int k0) {  // W[k0..+63][n0..+127] transposed
#pragma unroll
        for (int s = tid; s < 2048; s += 256) {
            const int kk = s >> 5;
            const int n4 = (s & 31) << 2;
            const float4 f = *(const float4*)(W + (size_t)(k0 + kk) * EMBED + n0 + n4);
            buf[(n4 + 0) * LDS_STRIDE + kk] = f2bf(f.x);
            buf[(n4 + 1) * LDS_STRIDE + kk] = f2bf(f.y);
            buf[(n4 + 2) * LDS_STRIDE + kk] = f2bf(f.z);
            buf[(n4 + 3) * LDS_STRIDE + kk] = f2bf(f.w);
        }
    };

    const v8f vzero = {0.f, 0.f, 0.f, 0.f, 0.f, 0.f, 0.f, 0.f};
    v8f acc[4][2];
#pragma unroll
    for (int mi = 0; mi < 4; ++mi)
#pragma unroll
        for (int ni = 0; ni < 2; ++ni) acc[mi][ni] = vzero;

    stageA(lds_A[0], 0);
    stageB(lds_B[0], 0);
    for (int k0 = 0; k0 < EMBED; k0 += 64) {
        const int cur = (k0 >> 6) & 1;
        __syncthreads();
        if (k0 + 64 < EMBED) { stageA(lds_A[cur ^ 1], k0 + 64); stageB(lds_B[cur ^ 1], k0 + 64); }
        wmma_tile(lds_A[cur], lds_B[cur], wm, wn, col, kh, acc);
    }

#pragma unroll
    for (int mi = 0; mi < 4; ++mi)
#pragma unroll
        for (int ni = 0; ni < 2; ++ni) {
            const int cg = n0 + wn * 32 + ni * 16 + col;
            const float bb = bias[cg];
            Acc8 a; a.v = acc[mi][ni];
            const int rbase = m0 + wm * 64 + mi * 16 + kh * 8;
#pragma unroll
            for (int r = 0; r < 8; ++r)
                Y[(size_t)(rbase + r) * EMBED + cg] = f2bf(a.f[r] + bb);
        }
}

// ---------------------------------------------------------------------------
// S[b][i][j] = scale * sum_d Q[b][i][d]*K[b][j][d]  (bf16 in, fp32 out)
// K row-major IS the WMMA B operand (lane=column j, contiguous d): no transpose.
// Tiles staged with async global->LDS copies (ASYNCcnt).
__global__ __launch_bounds__(256)
void qk_kernel(const unsigned short* __restrict__ Q, const unsigned short* __restrict__ K,
               float* __restrict__ S) {
    __shared__ unsigned short lds_A[2][128 * LDS_STRIDE];
    __shared__ unsigned short lds_B[2][128 * LDS_STRIDE];

    const int tid  = threadIdx.x;
    const int j0   = blockIdx.x * 128;
    const int i0   = blockIdx.y * 128;
    const int b    = blockIdx.z;
    const unsigned short* Qb = Q + (size_t)b * SEQ_I * EMBED;
    const unsigned short* Kb = K + (size_t)b * SEQ_J * EMBED;
    float* Sb = S + (size_t)b * SEQ_I * SEQ_J;

    const int w    = tid >> 5;
    const int lane = tid & 31;
    const int wm   = w >> 2;
    const int wn   = w & 3;
    const int col  = lane & 15;
    const int kh   = lane >> 4;

    auto stage = [&](unsigned short* bufA, unsigned short* bufB, int d0) {
#pragma unroll
        for (int s = tid; s < 1024; s += 256) {      // 128 rows x 8 x 16B per matrix
            const int row = s >> 3;
            const int seg = (s & 7) << 3;            // bf16 index, 16B granules
            async_copy_b128(lds_off(&bufA[row * LDS_STRIDE + seg]),
                            Qb + (size_t)(i0 + row) * EMBED + d0 + seg);
            async_copy_b128(lds_off(&bufB[row * LDS_STRIDE + seg]),
                            Kb + (size_t)(j0 + row) * EMBED + d0 + seg);
        }
    };

    const v8f vzero = {0.f, 0.f, 0.f, 0.f, 0.f, 0.f, 0.f, 0.f};
    v8f acc[4][2];
#pragma unroll
    for (int mi = 0; mi < 4; ++mi)
#pragma unroll
        for (int ni = 0; ni < 2; ++ni) acc[mi][ni] = vzero;

    stage(lds_A[0], lds_B[0], 0);
    for (int d0 = 0; d0 < EMBED; d0 += 64) {
        const int cur = (d0 >> 6) & 1;
        wait_asynccnt0();      // this wave's async copies into buf[cur] landed
        __syncthreads();       // every wave's copies landed
        if (d0 + 64 < EMBED) stage(lds_A[cur ^ 1], lds_B[cur ^ 1], d0 + 64);
        wmma_tile(lds_A[cur], lds_B[cur], wm, wn, col, kh, acc);
    }

#pragma unroll
    for (int mi = 0; mi < 4; ++mi)
#pragma unroll
        for (int ni = 0; ni < 2; ++ni) {
            const int jg    = j0 + wn * 32 + ni * 16 + col;
            const int ibase = i0 + wm * 64 + mi * 16 + kh * 8;
            Acc8 a; a.v = acc[mi][ni];
#pragma unroll
            for (int r = 0; r < 8; ++r)
                Sb[(size_t)(ibase + r) * SEQ_J + jg] = a.f[r] * QK_SCALE;
        }
}

// ---------------------------------------------------------------------------
// In-place row softmax over SEQ_J=2048 fp32; one block per row.
__global__ __launch_bounds__(256)
void softmax_kernel(float* __restrict__ S) {
    __shared__ float red[256];
    float* row = S + (size_t)blockIdx.x * SEQ_J;
    const int tid = threadIdx.x;
    float4* r4 = (float4*)row;
    float4 x0 = r4[tid];
    float4 x1 = r4[tid + 256];

    float m = fmaxf(fmaxf(fmaxf(x0.x, x0.y), fmaxf(x0.z, x0.w)),
                    fmaxf(fmaxf(x1.x, x1.y), fmaxf(x1.z, x1.w)));
    red[tid] = m; __syncthreads();
#pragma unroll
    for (int s = 128; s > 0; s >>= 1) {
        if (tid < s) red[tid] = fmaxf(red[tid], red[tid + s]);
        __syncthreads();
    }
    m = red[0]; __syncthreads();

    x0.x = __expf(x0.x - m); x0.y = __expf(x0.y - m);
    x0.z = __expf(x0.z - m); x0.w = __expf(x0.w - m);
    x1.x = __expf(x1.x - m); x1.y = __expf(x1.y - m);
    x1.z = __expf(x1.z - m); x1.w = __expf(x1.w - m);

    red[tid] = (x0.x + x0.y + x0.z + x0.w) + (x1.x + x1.y + x1.z + x1.w);
    __syncthreads();
#pragma unroll
    for (int s = 128; s > 0; s >>= 1) {
        if (tid < s) red[tid] += red[tid + s];
        __syncthreads();
    }
    const float inv = 1.0f / red[0];

    x0.x *= inv; x0.y *= inv; x0.z *= inv; x0.w *= inv;
    x1.x *= inv; x1.y *= inv; x1.z *= inv; x1.w *= inv;
    r4[tid] = x0; r4[tid + 256] = x1;
}

// ---------------------------------------------------------------------------
// O[b][i][d] = sum_j P[b][i][j]*V[b][j][d]  (P fp32->bf16, V bf16, O fp32)
__global__ __launch_bounds__(256)
void pv_kernel(const float* __restrict__ P, const unsigned short* __restrict__ V,
               float* __restrict__ O) {
    __shared__ unsigned short lds_A[2][128 * LDS_STRIDE];
    __shared__ unsigned short lds_B[2][128 * LDS_STRIDE];

    const int tid  = threadIdx.x;
    const int d0   = blockIdx.x * 128;
    const int i0   = blockIdx.y * 128;
    const int b    = blockIdx.z;
    const float* Pb = P + (size_t)b * SEQ_I * SEQ_J;
    const unsigned short* Vb = V + (size_t)b * SEQ_J * EMBED;
    float* Ob = O + (size_t)b * SEQ_I * EMBED;

    const int w    = tid >> 5;
    const int lane = tid & 31;
    const int wm   = w >> 2;
    const int wn   = w & 3;
    const int col  = lane & 15;
    const int kh   = lane >> 4;

    auto stageA = [&](unsigned short* buf, int j0) {  // P[i0..+127][j0..+63] fp32->bf16
#pragma unroll
        for (int s = tid; s < 2048; s += 256) {
            const int row = s >> 4;
            const int c4  = (s & 15) << 2;
            const float4 f = *(const float4*)(Pb + (size_t)(i0 + row) * SEQ_J + j0 + c4);
            unsigned short* p = &buf[row * LDS_STRIDE + c4];
            p[0] = f2bf(f.x); p[1] = f2bf(f.y); p[2] = f2bf(f.z); p[3] = f2bf(f.w);
        }
    };
    auto stageB = [&](unsigned short* buf, int j0) {  // V[j0..+63][d0..+127] transposed
#pragma unroll
        for (int s = tid; s < 1024; s += 256) {
            const int kk = s >> 4;
            const int n8 = (s & 15) << 3;
            const uint4 u = *(const uint4*)(Vb + (size_t)(j0 + kk) * EMBED + d0 + n8);
            const unsigned short* us = (const unsigned short*)&u;
#pragma unroll
            for (int e = 0; e < 8; ++e)
                buf[(n8 + e) * LDS_STRIDE + kk] = us[e];
        }
    };

    const v8f vzero = {0.f, 0.f, 0.f, 0.f, 0.f, 0.f, 0.f, 0.f};
    v8f acc[4][2];
#pragma unroll
    for (int mi = 0; mi < 4; ++mi)
#pragma unroll
        for (int ni = 0; ni < 2; ++ni) acc[mi][ni] = vzero;

    stageA(lds_A[0], 0);
    stageB(lds_B[0], 0);
    for (int j0 = 0; j0 < SEQ_J; j0 += 64) {
        const int cur = (j0 >> 6) & 1;
        __syncthreads();
        if (j0 + 64 < SEQ_J) { stageA(lds_A[cur ^ 1], j0 + 64); stageB(lds_B[cur ^ 1], j0 + 64); }
        wmma_tile(lds_A[cur], lds_B[cur], wm, wn, col, kh, acc);
    }

#pragma unroll
    for (int mi = 0; mi < 4; ++mi)
#pragma unroll
        for (int ni = 0; ni < 2; ++ni) {
            const int dg    = d0 + wn * 32 + ni * 16 + col;
            const int ibase = i0 + wm * 64 + mi * 16 + kh * 8;
            Acc8 a; a.v = acc[mi][ni];
#pragma unroll
            for (int r = 0; r < 8; ++r)
                Ob[(size_t)(ibase + r) * EMBED + dg] = a.f[r];
        }
}

// ---------------------------------------------------------------------------
extern "C" void kernel_launch(void* const* d_in, const int* in_sizes, int n_in,
                              void* d_out, int out_size, void* d_ws, size_t ws_size,
                              hipStream_t stream) {
    const float* q_in = (const float*)d_in[0];
    const float* k_v  = (const float*)d_in[1];
    const float* Wq   = (const float*)d_in[2];
    const float* bq   = (const float*)d_in[3];
    const float* Wk   = (const float*)d_in[4];
    const float* bk   = (const float*)d_in[5];
    const float* Wv   = (const float*)d_in[6];
    const float* bv   = (const float*)d_in[7];
    float* out = (float*)d_out;

    // workspace: q | k | v (bf16) | S (fp32)   -> 32+32+32+128 MB = 224 MB
    const size_t qkvElems = (size_t)BATCH * SEQ_I * EMBED;  // 16M
    unsigned short* qb = (unsigned short*)d_ws;
    unsigned short* kb = qb + qkvElems;
    unsigned short* vb = kb + qkvElems;
    float* S = (float*)(vb + qkvElems);

    const dim3 blk(256);
    proj_kernel<<<dim3(EMBED / 128, (BATCH * SEQ_I) / 128), blk, 0, stream>>>(q_in, Wq, bq, qb);
    proj_kernel<<<dim3(EMBED / 128, (BATCH * SEQ_J) / 128), blk, 0, stream>>>(k_v, Wk, bk, kb);
    proj_kernel<<<dim3(EMBED / 128, (BATCH * SEQ_J) / 128), blk, 0, stream>>>(k_v, Wv, bv, vb);
    qk_kernel<<<dim3(SEQ_J / 128, SEQ_I / 128, BATCH), blk, 0, stream>>>(qb, kb, S);
    softmax_kernel<<<dim3(BATCH * SEQ_I), blk, 0, stream>>>(S);
    pv_kernel<<<dim3(EMBED / 128, SEQ_I / 128, BATCH), blk, 0, stream>>>(S, vb, out);
}
